// DeepGate3_40776419508348
// MI455X (gfx1250) — matrix operand order repaired and checked by reading, not compiled, not run on hardware
//
#include <hip/hip_runtime.h>

// ---------------------------------------------------------------------------
// CDNA5 (gfx1250) DeepGate3 forward.  All dense GEMMs via
// v_wmma_f32_16x16x32_bf16 (fp32 accumulate).  wave32, 256-thread blocks.
// Weights pre-packed into WMMA B-fragment tiles (direct coalesced global
// loads, L2-resident) -> barrier-free GEMM k-loops.  Encoder activations are
// bf16 end-to-end and staged into LDS with async global->LDS copies.
// ---------------------------------------------------------------------------

typedef __attribute__((ext_vector_type(16))) __bf16 v16bf;
typedef __attribute__((ext_vector_type(8)))  float  v8f;
typedef int v4i_vec __attribute__((__vector_size__(4 * sizeof(int))));

#define NN      400000
#define NHOP    4096
#define NPAIR   262144
#define NPATH   512
#define SCALE_F 0.17677669529663687f   // 1/sqrt(32)

#if defined(__has_builtin)
#  if __has_builtin(__builtin_amdgcn_global_load_async_to_lds_b128) && \
      __has_builtin(__builtin_amdgcn_s_wait_asynccnt)
#    define HAVE_ASYNC_LDS 1
#  endif
#endif
#ifndef HAVE_ASYNC_LDS
#  define HAVE_ASYNC_LDS 0
#endif

__device__ __forceinline__ unsigned short f2bf(float f) {
    union { float f; unsigned u; } c; c.f = f;
    unsigned u = c.u;
    u += 0x7FFFu + ((u >> 16) & 1u);       // round-to-nearest-even
    return (unsigned short)(u >> 16);
}
__device__ __forceinline__ float bf2f(unsigned short h) {
    union { unsigned u; float f; } c; c.u = ((unsigned)h) << 16;
    return c.f;
}
__device__ __forceinline__ v8f vzero() {
    v8f z;
#pragma unroll
    for (int i = 0; i < 8; ++i) z[i] = 0.f;
    return z;
}
__device__ __forceinline__ v8f wmma_bf16(v16bf a, v16bf b, v8f c) {
    return __builtin_amdgcn_wmma_f32_16x16x32_bf16(false, a, false, b,
                                                   (short)0, c, false, false);
}

// 16-byte global(bf16) -> LDS copy; async path on gfx1250 (ASYNCcnt).
__device__ __forceinline__ void cp16_g2l(const unsigned short* gsrc,
                                         unsigned short* ldst) {
#if HAVE_ASYNC_LDS
    __builtin_amdgcn_global_load_async_to_lds_b128(
        (__attribute__((address_space(1))) v4i_vec*)(unsigned short*)gsrc,
        (__attribute__((address_space(3))) v4i_vec*)ldst, 0, 0);
#else
    *reinterpret_cast<uint4*>(ldst) = *reinterpret_cast<const uint4*>(gsrc);
#endif
}
__device__ __forceinline__ void async_fence() {
#if HAVE_ASYNC_LDS
    __builtin_amdgcn_s_wait_asynccnt(0);
#endif
}

// A fragment: 16x32 bf16 tile from LDS, row-major, stride AS (elements).
__device__ __forceinline__ v16bf load_a_frag(const unsigned short* lds, int AS,
                                             int mbase, int kbase, int lane) {
    const unsigned short* p = lds + (size_t)(mbase + (lane & 15)) * AS
                                  + kbase + ((lane >> 4) & 1) * 8;
    union { v16bf v; uint4 u[2]; } r;
    r.u[0] = *reinterpret_cast<const uint4*>(p);
    r.u[1] = *reinterpret_cast<const uint4*>(p + 16);
    return r.v;
}
// B fragment straight from pre-packed global weights.
// Packed tile (kt,nt) = 64 uint4: [half h][lane][8 bf16], fully coalesced.
__device__ __forceinline__ v16bf load_b_packed(const unsigned short* Wp, int KT,
                                               int ktAbs, int nt, int lane) {
    const uint4* p = reinterpret_cast<const uint4*>(Wp)
                   + ((size_t)(nt * KT + ktAbs) * 64 + lane);
    union { v16bf v; uint4 u[2]; } r;
    r.u[0] = p[0];
    r.u[1] = p[32];
    return r.v;
}

// Barrier-free block GEMM: NACC 16x16 tiles per wave, K stepped by 32.
// A in LDS bf16 [rows][AS]; W packed (KT = total K tiles), tiles kt0+.., cols nt0+j.
template<int NACC>
__device__ __forceinline__ void gemm_block(const unsigned short* Alds, int AS, int KD,
                                           const unsigned short* Wp, int KT, int kt0,
                                           v8f* acc, int mtile, int nt0) {
    const int lane = threadIdx.x & 31;
    for (int k0 = 0, kt = 0; k0 < KD; k0 += 32, ++kt) {
        v16bf a = load_a_frag(Alds, AS, mtile * 16, k0, lane);
#pragma unroll
        for (int j = 0; j < NACC; ++j) {
            v16bf b = load_b_packed(Wp, KT, kt0 + kt, nt0 + j, lane);
            acc[j] = wmma_bf16(a, b, acc[j]);
        }
    }
}

// C/D store: VGPR i -> row mtile*16 + i + 8*(lane>=16), col nt*16+(lane&15)
template<int NACC>
__device__ __forceinline__ void epi_store(unsigned short* dst, int DS, const v8f* acc,
                                          int mtile, int nt0, int lane,
                                          const float* bias, int boff,
                                          const unsigned short* resid, bool relu) {
    int col = lane & 15, g = (lane >> 4) & 1;
#pragma unroll
    for (int j = 0; j < NACC; ++j) {
        int n = (nt0 + j) * 16 + col;
        float bv = bias ? bias[boff + n] : 0.f;
#pragma unroll
        for (int i = 0; i < 8; ++i) {
            int r = mtile * 16 + g * 8 + i;
            float v = acc[j][i] + bv;
            if (resid) v += bf2f(resid[(size_t)r * DS + n]);
            if (relu) v = fmaxf(v, 0.f);
            dst[(size_t)r * DS + n] = f2bf(v);
        }
    }
}

// fp32 [K][N] -> bf16 packed WMMA-B tiles (tile order [nt][kt]).
__global__ void pack_b_kernel(const float* W, unsigned short* out, int K, int N) {
    int i = blockIdx.x * 256 + threadIdx.x;
    if (i >= K * N) return;
    int e4   = i & 7;
    int lane = (i >> 3) & 31;
    int h    = (i >> 8) & 1;
    int tile = i >> 9;
    int KT = K >> 5;
    int kt = tile % KT, nt = tile / KT;
    int kk = kt * 32 + h * 16 + ((lane >> 4) & 1) * 8 + e4;
    int nn = nt * 16 + (lane & 15);
    out[i] = f2bf(W[(size_t)kk * N + nn]);
}

// ---------------------------------------------------------------------------
// fused 3-layer MLP: y = (relu(LN(relu(LN(X@W1+b1))@W2+b2)))@W3+b3
// 32 rows / block, dh = 256, din in {128,256}, dout <= 4.
// mode 0: X = x[row]  mode 1: interleave(base[idx0],base[idx1])  mode 2: concat
// ---------------------------------------------------------------------------
struct MlpArgs {
    const float* x;
    const int*   idx0; const int* idx1;
    const float* base; const float* aux;
    const unsigned short* w1; const float* b1; const float* g1; const float* be1;
    const unsigned short* w2; const float* b2; const float* g2; const float* be2;
    const float* w3; const float* b3;
    float* out;
    int M, din, dout, mode;
};

__global__ __launch_bounds__(256) void fused_mlp3_kernel(MlpArgs A) {
    __shared__ __align__(16) unsigned short XB[32 * 264];   // X / pre-LN staging
    __shared__ __align__(16) unsigned short HB[32 * 264];   // hidden (post LN+relu)
    const int tid = threadIdx.x, w = tid >> 5, lane = tid & 31;
    const int r0 = blockIdx.x * 32;
    const int din = A.din;

    for (int idx = tid; idx < 32 * din; idx += 256) {
        int r = idx / din, f = idx - r * din;
        int row = r0 + r;
        float v = 0.f;
        if (row < A.M) {
            if (A.mode == 0)       v = A.x[(size_t)row * din + f];
            else if (A.mode == 1) {
                int gi = (f & 1) ? A.idx1[row] : A.idx0[row];
                v = A.base[(size_t)gi * 128 + (f >> 1)];
            } else {
                v = (f < 128) ? A.base[(size_t)A.idx0[row] * 128 + f]
                              : A.aux[(size_t)row * 128 + (f - 128)];
            }
        }
        XB[r * 264 + f] = f2bf(v);
    }
    __syncthreads();

    const int mtile = w & 1, nt0 = (w >> 1) * 4;   // 32x256 = 32 tiles / 8 waves
    v8f acc[4];

    // --- layer 1 ---
    for (int j = 0; j < 4; ++j) acc[j] = vzero();
    gemm_block<4>(XB, 264, din, A.w1, din >> 5, 0, acc, mtile, nt0);
    __syncthreads();
    epi_store<4>(XB, 264, acc, mtile, nt0, lane, A.b1, 0, nullptr, false);
    __syncthreads();
    if (tid < 32) {                                 // LN + relu -> HB
        int r = tid; float m = 0.f, vv = 0.f;
        for (int c = 0; c < 256; ++c) m += bf2f(XB[r * 264 + c]);
        m *= (1.f / 256.f);
        for (int c = 0; c < 256; ++c) { float d = bf2f(XB[r * 264 + c]) - m; vv += d * d; }
        float inv = rsqrtf(vv * (1.f / 256.f) + 1e-5f);
        for (int c = 0; c < 256; ++c) {
            float y = (bf2f(XB[r * 264 + c]) - m) * inv * A.g1[c] + A.be1[c];
            HB[r * 264 + c] = f2bf(fmaxf(y, 0.f));
        }
    }
    __syncthreads();
    // --- layer 2 ---
    for (int j = 0; j < 4; ++j) acc[j] = vzero();
    gemm_block<4>(HB, 264, 256, A.w2, 8, 0, acc, mtile, nt0);
    __syncthreads();
    epi_store<4>(XB, 264, acc, mtile, nt0, lane, A.b2, 0, nullptr, false);
    __syncthreads();
    if (tid < 32) {
        int r = tid; float m = 0.f, vv = 0.f;
        for (int c = 0; c < 256; ++c) m += bf2f(XB[r * 264 + c]);
        m *= (1.f / 256.f);
        for (int c = 0; c < 256; ++c) { float d = bf2f(XB[r * 264 + c]) - m; vv += d * d; }
        float inv = rsqrtf(vv * (1.f / 256.f) + 1e-5f);
        for (int c = 0; c < 256; ++c) {
            float y = (bf2f(XB[r * 264 + c]) - m) * inv * A.g2[c] + A.be2[c];
            HB[r * 264 + c] = f2bf(fmaxf(y, 0.f));
        }
    }
    __syncthreads();
    // --- layer 3 (tiny, fp32 VALU) ---
    for (int idx = tid; idx < 32 * A.dout; idx += 256) {
        int r = idx / A.dout, o = idx - r * A.dout;
        int row = r0 + r;
        if (row >= A.M) continue;
        float s = A.b3[o];
        for (int k = 0; k < 256; ++k) s += bf2f(HB[r * 264 + k]) * A.w3[k * A.dout + o];
        A.out[(size_t)row * A.dout + o] = s;
    }
}

// ---------------------------------------------------------------------------
// Post-norm transformer encoder layer, d=128, 4 heads, hd=32, FF=2048, relu.
// One block per sequence; S in {16 (hops), 64 (paths)}.  Activations bf16.
// ---------------------------------------------------------------------------
struct EncArgs {
    const unsigned short* xin; unsigned short* xout;
    const unsigned short *wq, *wk, *wv, *wo, *wf1, *wf2;
    const float *bq, *bk, *bv, *bo, *ga1, *bb1, *ga2, *bb2, *bf1, *bf2;
    const int* stats;    // [nseq][8] for hop key-bias; null for pathformer
    int nseq;
};

template<int S>
__global__ __launch_bounds__(256) void enc_layer_kernel(EncArgs A) {
    constexpr int MT = S / 16;
    constexpr int STR = 136;
    __shared__ __align__(16) unsigned short XB[S * STR];   // x (bf16 residual)
    __shared__ __align__(16) unsigned short BA[S * STR];   // q -> k -> v -> preLN / ff h
    __shared__ __align__(16) unsigned short BB[S * STR];   // attn-out / pre-LN2
    __shared__ float SB[16];
    const int tid = threadIdx.x, w = tid >> 5, lane = tid & 31;
    const int seq = blockIdx.x;
    const unsigned short* xin = A.xin + (size_t)seq * S * 128;

    // async stage x tile (bf16, raw bytes) into LDS
    for (int c = tid; c < S * 16; c += 256) {
        int r = c >> 4, ch = c & 15;
        cp16_g2l(xin + (size_t)r * 128 + ch * 8, &XB[r * STR + ch * 8]);
    }
    if (tid < 16) {
        float b = 0.f;
        if constexpr (S == 16) {
            if (tid >= 10) b = -1e9f;
            else if (tid >= 1 && tid <= 8)
                if (A.stats[seq * 8 + tid - 1] == -1) b = -1e9f;
        }
        SB[tid] = b;
    }
    async_fence();
    __syncthreads();

    const int mtile = w % MT, nt0 = (w / MT) * MT;   // S*128 tiles / 8 waves
    v8f acc[MT];

    // ---- Q ----
    for (int j = 0; j < MT; ++j) acc[j] = vzero();
    gemm_block<MT>(XB, STR, 128, A.wq, 4, 0, acc, mtile, nt0);
    __syncthreads();
    epi_store<MT>(BA, STR, acc, mtile, nt0, lane, A.bq, 0, nullptr, false);
    __syncthreads();
    // cache q rows in registers: thread -> (head, query)
    float qreg[32]; float probs[S];
    const bool att = tid < 4 * S;
    const int ah = tid / S, aq = tid % S;
    if (att)
#pragma unroll
        for (int d = 0; d < 32; ++d) qreg[d] = bf2f(BA[aq * STR + ah * 32 + d]);
    // ---- K ---- (overwrites BA; q already cached, sync below covers copies)
    for (int j = 0; j < MT; ++j) acc[j] = vzero();
    gemm_block<MT>(XB, STR, 128, A.wk, 4, 0, acc, mtile, nt0);
    __syncthreads();
    epi_store<MT>(BA, STR, acc, mtile, nt0, lane, A.bk, 0, nullptr, false);
    __syncthreads();
    if (att) {                                   // scores + softmax (regs)
        float mx = -1e30f;
#pragma unroll 1
        for (int j = 0; j < S; ++j) {
            float s = 0.f;
            for (int d = 0; d < 32; ++d) s += qreg[d] * bf2f(BA[j * STR + ah * 32 + d]);
            s = s * SCALE_F + (S == 16 ? SB[j] : 0.f);
            probs[j] = s; mx = fmaxf(mx, s);
        }
        float den = 0.f;
        for (int j = 0; j < S; ++j) { probs[j] = __expf(probs[j] - mx); den += probs[j]; }
        float inv = 1.f / den;
        for (int j = 0; j < S; ++j) probs[j] *= inv;
    }
    // ---- V ---- (overwrites BA; scores done before the post-gemm sync)
    for (int j = 0; j < MT; ++j) acc[j] = vzero();
    gemm_block<MT>(XB, STR, 128, A.wv, 4, 0, acc, mtile, nt0);
    __syncthreads();
    epi_store<MT>(BA, STR, acc, mtile, nt0, lane, A.bv, 0, nullptr, false);
    __syncthreads();
    if (att) {                                   // attn @ V -> BB
        for (int d = 0; d < 32; ++d) {
            float o = 0.f;
            for (int j = 0; j < S; ++j) o += probs[j] * bf2f(BA[j * STR + ah * 32 + d]);
            BB[aq * STR + ah * 32 + d] = f2bf(o);
        }
    }
    __syncthreads();
    // ---- O proj + residual -> BA (pre-LN1) ----
    for (int j = 0; j < MT; ++j) acc[j] = vzero();
    gemm_block<MT>(BB, STR, 128, A.wo, 4, 0, acc, mtile, nt0);
    __syncthreads();
    epi_store<MT>(BA, STR, acc, mtile, nt0, lane, A.bo, 0, XB, false);
    __syncthreads();
    if (tid < S) {                               // LN1 -> new x in XB
        int r = tid; float m = 0.f, vv = 0.f;
        for (int c = 0; c < 128; ++c) m += bf2f(BA[r * STR + c]);
        m *= (1.f / 128.f);
        for (int c = 0; c < 128; ++c) { float d = bf2f(BA[r * STR + c]) - m; vv += d * d; }
        float inv = rsqrtf(vv * (1.f / 128.f) + 1e-5f);
        for (int c = 0; c < 128; ++c)
            XB[r * STR + c] = f2bf((bf2f(BA[r * STR + c]) - m) * inv * A.ga1[c] + A.bb1[c]);
    }
    __syncthreads();
    // ---- FF: 128 -> 2048 -> 128, chunked by 128 hidden cols, fused ----
    v8f facc[MT];
    for (int j = 0; j < MT; ++j) facc[j] = vzero();
    for (int c = 0; c < 16; ++c) {
        v8f h1[MT];
        for (int j = 0; j < MT; ++j) h1[j] = vzero();
        gemm_block<MT>(XB, STR, 128, A.wf1, 4, 0, h1, mtile, c * 8 + nt0);
        __syncthreads();                         // prev FF2 reads of BA done
        epi_store<MT>(BA, STR, h1, mtile, nt0, lane, A.bf1, c * 128, nullptr, true);
        __syncthreads();
        gemm_block<MT>(BA, STR, 128, A.wf2, 64, c * 4, facc, mtile, nt0);
    }
    __syncthreads();
    epi_store<MT>(BB, STR, facc, mtile, nt0, lane, A.bf2, 0, XB, false);
    __syncthreads();
    if (tid < S) {                               // LN2 -> global out (bf16)
        int r = tid; float m = 0.f, vv = 0.f;
        for (int c = 0; c < 128; ++c) m += bf2f(BB[r * STR + c]);
        m *= (1.f / 128.f);
        for (int c = 0; c < 128; ++c) { float d = bf2f(BB[r * STR + c]) - m; vv += d * d; }
        float inv = rsqrtf(vv * (1.f / 128.f) + 1e-5f);
        unsigned short* o = A.xout + (size_t)seq * S * 128 + (size_t)r * 128;
        for (int c = 0; c < 128; ++c)
            o[c] = f2bf((bf2f(BB[r * STR + c]) - m) * inv * A.ga2[c] + A.bb2[c]);
    }
}

// --------------------------- small helper kernels --------------------------
__global__ void build_hop_seq_kernel(const float* src, const int* pi, const int* stats,
                                     const int* po, const float* cls_tok,
                                     const float* zero_tok, const float* one_tok,
                                     const float* pos_emb, int use_tokens,
                                     unsigned short* out) {
    int h = blockIdx.x, f = threadIdx.x;     // block 128
    unsigned short* o = out + (size_t)h * 16 * 128;
    o[f] = f2bf(cls_tok[f] + pos_emb[f]);
    for (int j = 0; j < 8; ++j) {
        int s = stats[h * 8 + j];
        float v;
        if (s == -1)                    v = 0.f;
        else if (use_tokens && s == 0)  v = zero_tok[f];
        else if (use_tokens && s == 1)  v = one_tok[f];
        else                            v = src[(size_t)pi[h * 8 + j] * 128 + f];
        o[(1 + j) * 128 + f] = f2bf(v + pos_emb[(1 + j) * 128 + f]);
    }
    o[9 * 128 + f] = f2bf(src[(size_t)po[h] * 128 + f] + pos_emb[9 * 128 + f]);
    for (int r = 10; r < 16; ++r) o[r * 128 + f] = 0;
}

__global__ void build_path_seq_kernel(const float* hs, const int* paths, const int* plen,
                                      const float* clsp, const float* ppos,
                                      unsigned short* out) {
    int p = blockIdx.x, f = threadIdx.x;     // block 128
    unsigned short* o = out + (size_t)p * 64 * 128;
    o[f] = f2bf(clsp[f] + ppos[f]);
    int L = plen[p];
    for (int j = 0; j < 63; ++j) {
        float v = (j < L) ? hs[(size_t)paths[p * 63 + j] * 128 + f] : 0.f;
        o[(1 + j) * 128 + f] = f2bf(v + ppos[(1 + j) * 128 + f]);
    }
}

__global__ void extract_cls_kernel(const unsigned short* seqbuf, int S, float* out) {
    int i = blockIdx.x, f = threadIdx.x;
    out[(size_t)i * 128 + f] = bf2f(seqbuf[(size_t)i * S * 128 + f]);
}

// hop_tt head: LN(relu(x@W1+b1)) @ W2 + b2   (128->512->64), fp32 VALU
__global__ __launch_bounds__(256) void hop_head_kernel(const float* x, const float* w1,
        const float* b1, const float* g, const float* be, const float* w2,
        const float* b2, float* out) {
    __shared__ float xs[128];
    __shared__ float H[512];
    __shared__ float ps[256];
    int row = blockIdx.x, tid = threadIdx.x;
    for (int f = tid; f < 128; f += 256) xs[f] = x[(size_t)row * 128 + f];
    __syncthreads();
    for (int o = tid; o < 512; o += 256) {
        float s = b1[o];
        for (int k = 0; k < 128; ++k) s += xs[k] * w1[k * 512 + o];
        H[o] = fmaxf(s, 0.f);
    }
    __syncthreads();
    ps[tid] = H[tid] + H[tid + 256];
    __syncthreads();
    for (int st = 128; st > 0; st >>= 1) { if (tid < st) ps[tid] += ps[tid + st]; __syncthreads(); }
    float m = ps[0] * (1.f / 512.f);
    __syncthreads();
    float d0 = H[tid] - m, d1 = H[tid + 256] - m;
    ps[tid] = d0 * d0 + d1 * d1;
    __syncthreads();
    for (int st = 128; st > 0; st >>= 1) { if (tid < st) ps[tid] += ps[tid + st]; __syncthreads(); }
    float inv = rsqrtf(ps[0] * (1.f / 512.f) + 1e-5f);
    __syncthreads();
    H[tid]       = (H[tid]       - m) * inv * g[tid]       + be[tid];
    H[tid + 256] = (H[tid + 256] - m) * inv * g[tid + 256] + be[tid + 256];
    __syncthreads();
    if (tid < 64) {
        float s = b2[tid];
        for (int k = 0; k < 512; ++k) s += H[k] * w2[k * 64 + tid];
        out[(size_t)row * 64 + tid] = s;
    }
}

__global__ void gate_count_kernel(const int* paths, const int* plen, const int* gate,
                                  float* out1, float* out2) {
    __shared__ int c1, c2;
    int p = blockIdx.x, tid = threadIdx.x;   // block 64
    if (tid == 0) { c1 = 0; c2 = 0; }
    __syncthreads();
    if (tid < 63 && tid < plen[p]) {
        int g = gate[(size_t)paths[p * 63 + tid]];
        if (g == 1) atomicAdd(&c1, 1);
        else if (g == 2) atomicAdd(&c2, 1);
    }
    __syncthreads();
    if (tid == 0) { out1[p] = (float)c1; out2[p] = (float)c2; }
}

// ---------------------------------------------------------------------------
// host orchestration
// ---------------------------------------------------------------------------
extern "C" void kernel_launch(void* const* d_in, const int* in_sizes, int n_in,
                              void* d_out, int out_size, void* d_ws, size_t ws_size,
                              hipStream_t stream) {
    (void)in_sizes; (void)n_in; (void)out_size; (void)ws_size;
    enum { IN_HS = 0, IN_HF, IN_HOP_PI, IN_STATS, IN_HOP_PO, IN_CPI, IN_PATHS,
           IN_PLEN, IN_GATE, IN_ONG, IN_OUTG,
           IN_CLS = 11, IN_CLSP, IN_ZERO, IN_ONE, IN_DC, IN_POS, IN_PPOS };
    const int M_PROB = 18, M_LEVEL = 28, M_CONNECT = 38, M_ONPATH = 48,
              M_PLEN = 58, M_G1 = 68, M_G2 = 78, M_NUM = 88, M_HOPLEV = 98;
    const int HH = 108, TF0 = 114, PF0 = 162;

    auto FP = [&](int i) { return (const float*)d_in[i]; };
    auto IP = [&](int i) { return (const int*)d_in[i]; };

    char* wsb = (char*)d_ws;
    size_t off = 0;
    auto alloc = [&](size_t bytes) -> void* {
        off = (off + 255) & ~(size_t)255;
        void* p = wsb + off;
        off += bytes;
        return p;
    };
    // pack fp32 [K][N] weight into bf16 WMMA-B tiles
    auto pack = [&](int idx, int K, int N) -> const unsigned short* {
        unsigned short* dst = (unsigned short*)alloc((size_t)K * N * 2);
        pack_b_kernel<<<(unsigned)(((size_t)K * N + 255) / 256), 256, 0, stream>>>(
            FP(idx), dst, K, N);
        return dst;
    };

    struct EncW { const unsigned short *wq, *wk, *wv, *wo, *wf1, *wf2; };
    EncW enc[6];
    for (int L = 0; L < 6; ++L) {
        int b = (L < 3) ? (TF0 + 16 * L) : (PF0 + 16 * (L - 3));
        enc[L].wq  = pack(b + 0, 128, 128);
        enc[L].wk  = pack(b + 2, 128, 128);
        enc[L].wv  = pack(b + 4, 128, 128);
        enc[L].wo  = pack(b + 6, 128, 128);
        enc[L].wf1 = pack(b + 12, 128, 2048);
        enc[L].wf2 = pack(b + 14, 2048, 128);
    }
    struct MlpW { const unsigned short *w1, *w2; };
    const int mbase[9] = { M_PROB, M_LEVEL, M_CONNECT, M_ONPATH, M_PLEN, M_G1, M_G2, M_NUM, M_HOPLEV };
    const int mdin[9]  = { 128, 128, 256, 256, 128, 128, 128, 128, 128 };
    MlpW mw[9];
    for (int i = 0; i < 9; ++i) {
        mw[i].w1 = pack(mbase[i] + 0, mdin[i], 256);
        mw[i].w2 = pack(mbase[i] + 4, 256, 256);
    }

    // activation buffers (bf16 between encoder layers)
    unsigned short* bufA = (unsigned short*)alloc((size_t)NHOP * 16 * 128 * 2);
    unsigned short* bufB = (unsigned short*)alloc((size_t)NHOP * 16 * 128 * 2);
    float* clsC = (float*)alloc((size_t)NHOP * 128 * 4);
    float* clsD = (float*)alloc((size_t)NPATH * 128 * 4);

    float* out = (float*)d_out;
    float* o_prob   = out;
    float* o_hoptt  = out + 400000;
    float* o_level  = out + 662144;
    float* o_conn   = out + 1062144;
    float* o_on     = out + 1848576;
    float* o_out    = out + 1849088;
    float* o_plen   = out + 1849600;
    float* o_pg1    = out + 1850112;
    float* o_pg2    = out + 1850624;
    float* o_num    = out + 1851136;
    float* o_hoplev = out + 1855232;
    float* o_g1     = out + 1859328;
    float* o_g2     = out + 1859840;

    auto run_mlp = [&](int mi, int M, int dout, int mode, const float* x,
                       const int* i0, const int* i1, const float* base,
                       const float* aux, float* outp) {
        int b = mbase[mi];
        MlpArgs a{};
        a.x = x; a.idx0 = i0; a.idx1 = i1; a.base = base; a.aux = aux;
        a.w1 = mw[mi].w1; a.b1 = FP(b + 1); a.g1 = FP(b + 2); a.be1 = FP(b + 3);
        a.w2 = mw[mi].w2; a.b2 = FP(b + 5); a.g2 = FP(b + 6); a.be2 = FP(b + 7);
        a.w3 = FP(b + 8); a.b3 = FP(b + 9);
        a.out = outp; a.M = M; a.din = mdin[mi]; a.dout = dout; a.mode = mode;
        fused_mlp3_kernel<<<(unsigned)((M + 31) / 32), 256, 0, stream>>>(a);
    };
    auto run_enc = [&](int L, int S, const unsigned short* xin, unsigned short* xout,
                       const int* stats, int nseq) {
        int b = (L < 3) ? (TF0 + 16 * L) : (PF0 + 16 * (L - 3));
        EncArgs e{};
        e.xin = xin; e.xout = xout;
        e.wq = enc[L].wq; e.wk = enc[L].wk; e.wv = enc[L].wv; e.wo = enc[L].wo;
        e.wf1 = enc[L].wf1; e.wf2 = enc[L].wf2;
        e.bq = FP(b + 1); e.bk = FP(b + 3); e.bv = FP(b + 5); e.bo = FP(b + 7);
        e.ga1 = FP(b + 8); e.bb1 = FP(b + 9); e.ga2 = FP(b + 10); e.bb2 = FP(b + 11);
        e.bf1 = FP(b + 13); e.bf2 = FP(b + 15);
        e.stats = stats; e.nseq = nseq;
        if (S == 16) enc_layer_kernel<16><<<nseq, 256, 0, stream>>>(e);
        else         enc_layer_kernel<64><<<nseq, 256, 0, stream>>>(e);
    };

    // ---- big node-wise MLPs ----
    run_mlp(0, NN, 1, 0, FP(IN_HF), nullptr, nullptr, nullptr, nullptr, o_prob);
    run_mlp(1, NN, 1, 0, FP(IN_HS), nullptr, nullptr, nullptr, nullptr, o_level);
    run_mlp(2, NPAIR, 3, 1, nullptr, IP(IN_CPI), IP(IN_CPI) + NPAIR, FP(IN_HS),
            nullptr, o_conn);

    // ---- hop pipeline, hf variant -> hop_tt ----
    build_hop_seq_kernel<<<NHOP, 128, 0, stream>>>(FP(IN_HF), IP(IN_HOP_PI), IP(IN_STATS),
            IP(IN_HOP_PO), FP(IN_CLS), FP(IN_ZERO), FP(IN_ONE), FP(IN_POS), 1, bufA);
    run_enc(0, 16, bufA, bufB, IP(IN_STATS), NHOP);
    run_enc(1, 16, bufB, bufA, IP(IN_STATS), NHOP);
    run_enc(2, 16, bufA, bufB, IP(IN_STATS), NHOP);
    extract_cls_kernel<<<NHOP, 128, 0, stream>>>(bufB, 16, clsC);
    hop_head_kernel<<<NHOP, 256, 0, stream>>>(clsC, FP(HH + 0), FP(HH + 1), FP(HH + 2),
            FP(HH + 3), FP(HH + 4), FP(HH + 5), o_hoptt);

    // ---- hop pipeline, hs variant -> readout_num / readout_hop_level ----
    build_hop_seq_kernel<<<NHOP, 128, 0, stream>>>(FP(IN_HS), IP(IN_HOP_PI), IP(IN_STATS),
            IP(IN_HOP_PO), FP(IN_CLS), FP(IN_ZERO), FP(IN_ONE), FP(IN_POS), 0, bufA);
    run_enc(0, 16, bufA, bufB, IP(IN_STATS), NHOP);
    run_enc(1, 16, bufB, bufA, IP(IN_STATS), NHOP);
    run_enc(2, 16, bufA, bufB, IP(IN_STATS), NHOP);
    extract_cls_kernel<<<NHOP, 128, 0, stream>>>(bufB, 16, clsC);
    run_mlp(7, NHOP, 1, 0, clsC, nullptr, nullptr, nullptr, nullptr, o_num);
    run_mlp(8, NHOP, 1, 0, clsC, nullptr, nullptr, nullptr, nullptr, o_hoplev);

    // ---- path pipeline ----
    build_path_seq_kernel<<<NPATH, 128, 0, stream>>>(FP(IN_HS), IP(IN_PATHS), IP(IN_PLEN),
            FP(IN_CLSP), FP(IN_PPOS), bufA);
    run_enc(3, 64, bufA, bufB, nullptr, NPATH);
    run_enc(4, 64, bufB, bufA, nullptr, NPATH);
    run_enc(5, 64, bufA, bufB, nullptr, NPATH);
    extract_cls_kernel<<<NPATH, 128, 0, stream>>>(bufB, 64, clsD);
    run_mlp(4, NPATH, 1, 0, clsD, nullptr, nullptr, nullptr, nullptr, o_plen);
    run_mlp(5, NPATH, 1, 0, clsD, nullptr, nullptr, nullptr, nullptr, o_pg1);
    run_mlp(6, NPATH, 1, 0, clsD, nullptr, nullptr, nullptr, nullptr, o_pg2);
    run_mlp(3, NPATH, 1, 2, nullptr, IP(IN_ONG), nullptr, FP(IN_HS), clsD, o_on);
    run_mlp(3, NPATH, 1, 2, nullptr, IP(IN_OUTG), nullptr, FP(IN_HS), clsD, o_out);

    // ---- gate counts ----
    gate_count_kernel<<<NPATH, 64, 0, stream>>>(IP(IN_PATHS), IP(IN_PLEN), IP(IN_GATE),
                                                o_g1, o_g2);
}
